// TL_CrossAttention_36679020708314
// MI455X (gfx1250) — matrix operand order repaired and checked by reading, not compiled, not run on hardware
//
#include <hip/hip_runtime.h>
#include <hip/hip_bf16.h>
#include <math.h>

// ---------------------------------------------------------------------------
// TL_CrossAttention for MI455X (gfx1250, wave32).
//
// Algebraic reformulation (exact):
//   q·k = (x Wq^T + bq)·kproj = x·(kproj @ Wq) + bq·kproj
// so the 154.6 GFLOP projection of the (16,128,64,768) query tensor collapses
// into a 1024x768x768 GEMM on the key side.  Remaining work is HBM-bound:
// 402 MB query read + 402 MB output write ~= 34.6 us at 23.3 TB/s.
//
// Pipeline (all on `stream`, workspace in d_ws):
//   G1: kproj = key   @ Wk^T + bk     (bf16 WMMA, NT)
//   G2: vproj = value @ Wv^T + bv     (bf16 WMMA, NT)
//   G3: kq    = kproj @ Wq            (bf16 WMMA, NN)
//   G4: sbias[m] = bq . kproj[m,:]
//   G5: scores = (query . kq + sbias)/sqrt(D); softmax over L; out = attn*vproj
//
// Round-1 fix: fp32->bf16 now uses the native convert (v_cvt_pk_bf16_f32)
// instead of an integer-ALU RNE emulation, cutting the VALU count in the
// WMMA inner loop by ~4x.
// ---------------------------------------------------------------------------

typedef __attribute__((ext_vector_type(16))) __bf16 v16bf;
typedef __attribute__((ext_vector_type(8)))  float  v8f;

#define WPB 8  // waves per block (256 threads, wave32)

static __device__ __forceinline__ __bf16 f2bf(float f) {
  return (__bf16)f;  // hardware cvt, RNE
}

// C[m,n] = sum_k A[m,k] * Bw(k,n) + bias[n]
//   BT=true : Bw(k,n) = W[n*K + k]   (i.e. A @ W^T, contiguous along k)
//   BT=false: Bw(k,n) = W[k*N + n]   (i.e. A @ W)
// One wave computes one 16x16 tile of C; K stepped by 32 per WMMA.
template <bool BT>
__global__ __launch_bounds__(256) void gemm_wmma_bf16(
    const float* __restrict__ A, const float* __restrict__ W,
    const float* __restrict__ bias, float* __restrict__ C,
    int M, int N, int K) {
  const int lane = threadIdx.x & 31;
  const int wave = threadIdx.x >> 5;
  const int tilesN = N >> 4;
  const int gid = blockIdx.x * WPB + wave;
  const int tm = (gid / tilesN) << 4;
  const int tn = (gid % tilesN) << 4;
  if (tm >= M) return;  // wave-uniform

  const int mrow = tm + (lane & 15);
  const int ncol = tn + (lane & 15);
  const int hi = lane >> 4;  // lane group 0/1

  v8f acc = {};
  for (int k0 = 0; k0 < K; k0 += 32) {
    // ---- A fragment (16x32 bf16): lane holds K = base..base+15 (16 consecutive)
    const float* arow = A + (size_t)mrow * K + (k0 + hi * 16);
    float4 a0 = *(const float4*)(arow + 0);
    float4 a1 = *(const float4*)(arow + 4);
    float4 a2 = *(const float4*)(arow + 8);
    float4 a3 = *(const float4*)(arow + 12);
    v16bf af;
    af[0] = f2bf(a0.x); af[1] = f2bf(a0.y); af[2]  = f2bf(a0.z); af[3]  = f2bf(a0.w);
    af[4] = f2bf(a1.x); af[5] = f2bf(a1.y); af[6]  = f2bf(a1.z); af[7]  = f2bf(a1.w);
    af[8] = f2bf(a2.x); af[9] = f2bf(a2.y); af[10] = f2bf(a2.z); af[11] = f2bf(a2.w);
    af[12]= f2bf(a3.x); af[13]= f2bf(a3.y); af[14] = f2bf(a3.z); af[15] = f2bf(a3.w);

    // ---- B fragment (32x16 bf16): lane = column n, 16 consecutive K values
    const int kb = k0 + hi * 16;
    v16bf bfr;
    if (BT) {
      const float* brow = W + (size_t)ncol * K + kb;
      float4 b0 = *(const float4*)(brow + 0);
      float4 b1 = *(const float4*)(brow + 4);
      float4 b2 = *(const float4*)(brow + 8);
      float4 b3 = *(const float4*)(brow + 12);
      bfr[0] = f2bf(b0.x); bfr[1] = f2bf(b0.y); bfr[2]  = f2bf(b0.z); bfr[3]  = f2bf(b0.w);
      bfr[4] = f2bf(b1.x); bfr[5] = f2bf(b1.y); bfr[6]  = f2bf(b1.z); bfr[7]  = f2bf(b1.w);
      bfr[8] = f2bf(b2.x); bfr[9] = f2bf(b2.y); bfr[10] = f2bf(b2.z); bfr[11] = f2bf(b2.w);
      bfr[12]= f2bf(b3.x); bfr[13]= f2bf(b3.y); bfr[14] = f2bf(b3.z); bfr[15] = f2bf(b3.w);
    } else {
#pragma unroll
      for (int e = 0; e < 16; ++e)
        bfr[e] = f2bf(W[(size_t)(kb + e) * N + ncol]);
    }

    acc = __builtin_amdgcn_wmma_f32_16x16x32_bf16(
        /*neg_a=*/false, af, /*neg_b=*/false, bfr,
        /*c_mod=*/(short)0, acc, /*reuse_a=*/false, /*reuse_b=*/false);
  }

  const float bn = bias ? bias[ncol] : 0.0f;
#pragma unroll
  for (int r = 0; r < 8; ++r) {  // VGPR r holds row r (lanes 0-15) / r+8 (16-31)
    const int row = tm + r + hi * 8;
    C[(size_t)row * N + ncol] = acc[r] + bn;
  }
}

// sbias[m] = dot(bq, kproj[m, :]),  D = 768, one wave per row
__global__ __launch_bounds__(256) void bias_dot_kernel(
    const float* __restrict__ kproj, const float* __restrict__ bq,
    float* __restrict__ sbias, int M) {
  const int lane = threadIdx.x & 31;
  const int wave = threadIdx.x >> 5;
  const int m = blockIdx.x * WPB + wave;
  if (m >= M) return;
  const float* row = kproj + (size_t)m * 768;
  float acc = 0.0f;
#pragma unroll
  for (int i = 0; i < 6; ++i) {
    const int d = i * 128 + lane * 4;
    float4 a = *(const float4*)(row + d);
    float4 b = *(const float4*)(bq + d);
    acc += a.x * b.x + a.y * b.y + a.z * b.z + a.w * b.w;
  }
#pragma unroll
  for (int off = 16; off; off >>= 1) acc += __shfl_xor(acc, off, 32);
  if (lane == 0) sbias[m] = acc;
}

// One block per (b,t): 64 dots over D=768, softmax over L=64, scale vproj.
__global__ __launch_bounds__(256) void attn_kernel(
    const float* __restrict__ query, const float* __restrict__ kq,
    const float* __restrict__ vproj, const float* __restrict__ sbias,
    float* __restrict__ out) {
  constexpr int T = 128, L = 64, D = 768;
  const float rs = 0.03608439182435161f;  // 1/sqrt(768)
  const int bt = blockIdx.x;
  const int b = bt / T;
  const int lane = threadIdx.x & 31;
  const int wave = threadIdx.x >> 5;
  __shared__ float sSc[L];

  const float* qbase = query + (size_t)bt * L * D;
  const float* kqb   = kq    + (size_t)b  * L * D;

#pragma unroll
  for (int li = 0; li < 8; ++li) {
    const int l = wave * 8 + li;
    const float* q  = qbase + (size_t)l * D;
    const float* kr = kqb   + (size_t)l * D;
    if (li + 1 < 8) __builtin_prefetch(q + D, 0, 0);  // global_prefetch_b8
    float acc = 0.0f;
#pragma unroll
    for (int i = 0; i < 6; ++i) {
      const int d = i * 128 + lane * 4;
      float4 a = *(const float4*)(q + d);
      float4 c = *(const float4*)(kr + d);
      acc += a.x * c.x + a.y * c.y + a.z * c.z + a.w * c.w;
    }
#pragma unroll
    for (int off = 16; off; off >>= 1) acc += __shfl_xor(acc, off, 32);
    if (lane == 0) sSc[l] = (acc + sbias[b * L + l]) * rs;
  }
  __syncthreads();

  // redundant per-thread softmax stats over 64 entries (cheap, LDS broadcast)
  float mx = -1e30f;
  for (int l = 0; l < L; ++l) mx = fmaxf(mx, sSc[l]);
  float den = 0.0f;
  for (int l = 0; l < L; ++l) den += __expf(sSc[l] - mx);
  const float inv = 1.0f / den;

  const float* vb = vproj + (size_t)b * L * D;
  float* ob = out + (size_t)bt * L * D;
#pragma unroll
  for (int li = 0; li < 8; ++li) {
    const int l = wave * 8 + li;
    const float p = __expf(sSc[l] - mx) * inv;
    const float* vr = vb + (size_t)l * D;
    float* orow = ob + (size_t)l * D;
#pragma unroll
    for (int i = 0; i < 6; ++i) {
      const int d = i * 128 + lane * 4;
      float4 v = *(const float4*)(vr + d);
      float4 o; o.x = p * v.x; o.y = p * v.y; o.z = p * v.z; o.w = p * v.w;
      *(float4*)(orow + d) = o;
    }
  }
}

extern "C" void kernel_launch(void* const* d_in, const int* in_sizes, int n_in,
                              void* d_out, int out_size, void* d_ws, size_t ws_size,
                              hipStream_t stream) {
  const float* query = (const float*)d_in[0];  // (16,128,64,768)
  const float* key   = (const float*)d_in[1];  // (16,64,768)
  const float* value = (const float*)d_in[2];  // (16,64,768)
  const float* Wq    = (const float*)d_in[3];  // (768,768)
  const float* bq    = (const float*)d_in[4];
  const float* Wk    = (const float*)d_in[5];
  const float* bk    = (const float*)d_in[6];
  const float* Wv    = (const float*)d_in[7];
  const float* bv    = (const float*)d_in[8];
  float* out = (float*)d_out;

  constexpr int M = 16 * 64;   // 1024 rows for key/value-side GEMMs
  constexpr int D = 768;
  constexpr size_t MAT = (size_t)M * D;  // 786432 floats

  float* ws    = (float*)d_ws;
  float* kproj = ws;            // 3 MB
  float* vproj = ws + MAT;      // 3 MB
  float* kq    = ws + 2 * MAT;  // 3 MB
  float* sbias = ws + 3 * MAT;  // 4 KB

  const dim3 blk(256);
  const int gemm_blocks = (M / 16) * (D / 16) / WPB;  // 384

  // kproj = key @ Wk^T + bk   |   vproj = value @ Wv^T + bv
  gemm_wmma_bf16<true><<<gemm_blocks, blk, 0, stream>>>(key,   Wk, bk, kproj, M, D, D);
  gemm_wmma_bf16<true><<<gemm_blocks, blk, 0, stream>>>(value, Wv, bv, vproj, M, D, D);
  // kq = kproj @ Wq  (bias already folded into kproj)
  gemm_wmma_bf16<false><<<gemm_blocks, blk, 0, stream>>>(kproj, Wq, nullptr, kq, M, D, D);
  // sbias = bq . kproj
  bias_dot_kernel<<<M / WPB, blk, 0, stream>>>(kproj, bq, sbias, M);
  // streaming attention pass: one block per (b,t)
  attn_kernel<<<16 * 128, blk, 0, stream>>>(query, kq, vproj, sbias, out);
}
